// GraphEncoder2_31782757991161
// MI455X (gfx1250) — matrix operand order repaired and checked by reading, not compiled, not run on hardware
//
#include <hip/hip_runtime.h>
#include <cstdint>
#include <cstddef>

#define NN 50000
#define EE 200000
#define CC 768
#define HH 256
#define OO 256
#define RR 5
#define TT 3
#define EL (EE + NN)   // edges incl. self loops

#if defined(__has_builtin)
#if __has_builtin(__builtin_amdgcn_global_load_async_to_lds_b128) && \
    __has_builtin(__builtin_amdgcn_s_wait_asynccnt)
#define HAVE_ASYNC_LDS 1
#endif
#endif
#ifndef HAVE_ASYNC_LDS
#define HAVE_ASYNC_LDS 0
#endif

typedef __attribute__((ext_vector_type(16))) __bf16 v16bf;
typedef __attribute__((ext_vector_type(8)))  float  v8f;
typedef __attribute__((ext_vector_type(4)))  int    v4i;
#if HAVE_ASYNC_LDS
typedef __attribute__((address_space(1))) v4i g_v4i;   // global
typedef __attribute__((address_space(3))) v4i l_v4i;   // LDS
#endif

static __device__ __forceinline__ unsigned short f2bf(float f) {
  unsigned int u = __float_as_uint(f);
  u += 0x7fffu + ((u >> 16) & 1u);            // round-to-nearest-even
  return (unsigned short)(u >> 16);
}
static __device__ __forceinline__ float lrelu(float z) { return z > 0.f ? z : 0.2f * z; }
static __device__ __forceinline__ void atomicMaxF(float* a, float v) {
  if (v >= 0.f) atomicMax((int*)a, __float_as_int(v));
  else          atomicMin((unsigned int*)a, __float_as_uint(v));
}
static __device__ __forceinline__ void edge_sd(const int* src, const int* dst, long e, int& s, int& d) {
  if (e < (long)EE) { s = src[e]; d = dst[e]; } else { s = d = (int)(e - EE); }
}

// ---------------- elementwise helpers ----------------
__global__ void k_fill(float* p, long n, float v) {
  long i = (long)blockIdx.x * blockDim.x + threadIdx.x;
  if (i < n) p[i] = v;
}
// relu(x + embed_node[node_type]) -> bf16 [N,C]
__global__ void k_node_prep(const float* __restrict__ x, const int* __restrict__ nt,
                            const float* __restrict__ emb, unsigned short* __restrict__ out) {
  long i = (long)blockIdx.x * blockDim.x + threadIdx.x;
  if (i < (long)NN * CC) {
    int c = (int)(i % CC); int n = (int)(i / CC);
    float v = x[i] + emb[(size_t)nt[n] * CC + c];
    out[i] = f2bf(v > 0.f ? v : 0.f);
  }
}
// Ebt[r,c] = relu(embed_edge[r]) @ fc_w[C:2C, c] + fc_b[c]   (R=5 rows, tiny)
__global__ void k_ebt(const float* __restrict__ embe, const float* __restrict__ fcw,
                      const float* __restrict__ fcb, float* __restrict__ ebt) {
  int i = blockIdx.x * blockDim.x + threadIdx.x;
  if (i < RR * CC) {
    int c = i % CC, r = i / CC;
    float acc = fcb[c];
    for (int k = 0; k < CC; ++k) {
      float e = embe[(size_t)r * CC + k];
      if (e > 0.f) acc += e * fcw[(size_t)(CC + k) * CC + c];
    }
    ebt[i] = acc;
  }
}

// ---------------- pack fp32 weight [K,Nw] into bf16 WMMA-fragment order ----------------
// Tile (nt, kt) -> base ((nt*KT + kt)*512); within tile lane = 16*half + (n&15) owns 16
// contiguous bf16: element j = B[kt*32 + 16*half + j, nt*16 + (n&15)].
__global__ void k_pack_b(const float* __restrict__ W, unsigned short* __restrict__ Bp,
                         int K, int Nw) {
  long i = (long)blockIdx.x * blockDim.x + threadIdx.x;
  if (i < (long)K * Nw) {
    int k = (int)(i / Nw), n = (int)(i % Nw);
    int kt = k >> 5, half = (k >> 4) & 1, j = k & 15;
    int nt = n >> 4, nl = n & 15;
    size_t dst = (((size_t)nt * (K >> 5) + kt) * 32 + half * 16 + nl) * (size_t)16 + j;
    Bp[dst] = f2bf(W[i]);
  }
}

// ---------------- WMMA GEMM: D[M,Nw] = A[M,K](bf16) @ Bpacked(bf16), fp32 accum --------
// block = 128 thr (4 waves). Wave w -> M-tile blockIdx.x*4+w; block covers N-tiles
// blockIdx.y*4..+3 whose packed B region (contiguous) is staged in LDS via async copy.
// A stream is double-buffered in registers (KT is even). K%32==0, Nw%64==0, M%16==0.
__global__ void k_wmma_gemm(const unsigned short* __restrict__ A,
                            const unsigned short* __restrict__ Bp,
                            float* __restrict__ D, int M, int K, int Nw) {
  extern __shared__ unsigned short Bs[];      // 4*KT*1024 bytes
  const int wave = threadIdx.x >> 5;
  const int lane = threadIdx.x & 31;
  const int tileM = blockIdx.x * 4 + wave;
  const bool active = (tileM * 16 < M);
  const int nt0 = blockIdx.y * 4;
  const int half = lane >> 4;
  const int mrow = lane & 15;
  const int KT = K >> 5;
  const size_t tstride = (size_t)KT * 512;    // elements between adjacent N-tiles

  // ---- cooperative async stage of B (4 contiguous N-tiles) into LDS ----
  const unsigned short* gsrc = Bp + (size_t)nt0 * tstride;
  const int nvec = KT * 256;                  // # of 16-byte chunks = 4*KT*512*2/16
  for (int i = threadIdx.x; i < nvec; i += blockDim.x) {
#if HAVE_ASYNC_LDS
    __builtin_amdgcn_global_load_async_to_lds_b128(
        (g_v4i*)(void*)(unsigned short*)(gsrc + (size_t)i * 8),
        (l_v4i*)(void*)(Bs + (size_t)i * 8), 0, 0);
#else
    *(uint4*)(void*)(Bs + (size_t)i * 8) = *(const uint4*)(const void*)(gsrc + (size_t)i * 8);
#endif
  }
#if HAVE_ASYNC_LDS
  __builtin_amdgcn_s_wait_asynccnt(0);
#endif
  __syncthreads();
  if (!active) return;                        // after the barrier; wave-uniform

  const unsigned short* ap = A + (size_t)(tileM * 16 + mrow) * K + 16 * half;
  const int lane16 = lane * 16;
  union AU { uint4 u[2]; v16bf v; };
  v8f acc0 = {}, acc1 = {}, acc2 = {}, acc3 = {};

  auto loadA = [&](AU& a, int ktc) {
    const unsigned short* p = ap + (size_t)ktc * 32;
    a.u[0] = *(const uint4*)(const void*)(p);
    a.u[1] = *(const uint4*)(const void*)(p + 8);
  };
  auto mma4 = [&](const AU& a, int ktc) {
    union BU { uint4 u[2]; v16bf v; } b0, b1, b2, b3;
    const unsigned short* q0 = Bs + (size_t)(0 * KT + ktc) * 512 + lane16;
    const unsigned short* q1 = Bs + (size_t)(1 * KT + ktc) * 512 + lane16;
    const unsigned short* q2 = Bs + (size_t)(2 * KT + ktc) * 512 + lane16;
    const unsigned short* q3 = Bs + (size_t)(3 * KT + ktc) * 512 + lane16;
    b0.u[0] = *(const uint4*)(const void*)q0; b0.u[1] = *(const uint4*)(const void*)(q0 + 8);
    b1.u[0] = *(const uint4*)(const void*)q1; b1.u[1] = *(const uint4*)(const void*)(q1 + 8);
    b2.u[0] = *(const uint4*)(const void*)q2; b2.u[1] = *(const uint4*)(const void*)(q2 + 8);
    b3.u[0] = *(const uint4*)(const void*)q3; b3.u[1] = *(const uint4*)(const void*)(q3 + 8);
    acc0 = __builtin_amdgcn_wmma_f32_16x16x32_bf16(false, a.v, false, b0.v, (short)0, acc0, false, false);
    acc1 = __builtin_amdgcn_wmma_f32_16x16x32_bf16(false, a.v, false, b1.v, (short)0, acc1, false, false);
    acc2 = __builtin_amdgcn_wmma_f32_16x16x32_bf16(false, a.v, false, b2.v, (short)0, acc2, false, false);
    acc3 = __builtin_amdgcn_wmma_f32_16x16x32_bf16(false, a.v, false, b3.v, (short)0, acc3, false, false);
  };

  AU a0, a1;
  loadA(a0, 0);
  int kt = 0;
  for (; kt + 2 < KT; kt += 2) {              // KT even (8 or 24)
    loadA(a1, kt + 1);
    mma4(a0, kt);
    loadA(a0, kt + 2);
    mma4(a1, kt + 1);
  }
  loadA(a1, KT - 1);
  mma4(a0, KT - 2);
  mma4(a1, KT - 1);

  // D layout per tile: VGPR v -> row = 16*tileM + v + 8*half, col = 16*nt + (lane&15)
  float* Drow = D + (size_t)(tileM * 16 + 8 * half) * Nw + (lane & 15);
  float* d0 = Drow + (size_t)(nt0 + 0) * 16;
  float* d1 = Drow + (size_t)(nt0 + 1) * 16;
  float* d2 = Drow + (size_t)(nt0 + 2) * 16;
  float* d3 = Drow + (size_t)(nt0 + 3) * 16;
#pragma unroll
  for (int v = 0; v < 8; ++v) {
    d0[(size_t)v * Nw] = acc0[v];
    d1[(size_t)v * Nw] = acc1[v];
    d2[(size_t)v * Nw] = acc2[v];
    d3[(size_t)v * Nw] = acc3[v];
  }
}

// ---------------- edge-MLP scatter-mean ----------------
__global__ void k_scatter_he(const int* __restrict__ src, const int* __restrict__ dst,
                             const int* __restrict__ et, const float* __restrict__ Y1n,
                             const float* __restrict__ ebt, float* __restrict__ accum,
                             float* __restrict__ deg) {
  int e = blockIdx.x;
  int s = src[e], d = dst[e], t = et[e];
  const float* ys = Y1n + (size_t)s * CC;
  const float* eb = ebt + (size_t)t * CC;
  float* ac = accum + (size_t)d * CC;
  for (int c = threadIdx.x; c < CC; c += blockDim.x) atomicAdd(&ac[c], ys[c] + eb[c]);
  if (threadIdx.x == 0) atomicAdd(&deg[d], 1.0f);
}
__global__ void k_xn(const float* __restrict__ accum, const float* __restrict__ deg,
                     unsigned short* __restrict__ xnbf) {
  long i = (long)blockIdx.x * blockDim.x + threadIdx.x;
  if (i < (long)NN * CC) {
    float dg = deg[i / CC]; dg = dg > 1.f ? dg : 1.f;
    xnbf[i] = f2bf(accum[i] / dg);
  }
}

// ---------------- GAT machinery ----------------
__global__ void k_scores(const float* __restrict__ Hm, const float* __restrict__ as,
                         const float* __restrict__ ad, float* __restrict__ ss,
                         float* __restrict__ sd, int F) {
  int n = blockIdx.x * blockDim.x + threadIdx.x;
  if (n < NN) {
    const float* h = Hm + (size_t)n * F;
    float a = 0.f, b = 0.f;
    for (int c = 0; c < F; ++c) { a += h[c] * as[c]; b += h[c] * ad[c]; }
    ss[n] = a; sd[n] = b;
  }
}
__global__ void k_logit_max(const int* src, const int* dst, const float* __restrict__ ss,
                            const float* __restrict__ sd, float* __restrict__ mx) {
  long e = (long)blockIdx.x * blockDim.x + threadIdx.x;
  if (e < (long)EL) { int s, d; edge_sd(src, dst, e, s, d); atomicMaxF(&mx[d], lrelu(ss[s] + sd[d])); }
}
__global__ void k_exp_sum(const int* src, const int* dst, const float* __restrict__ ss,
                          const float* __restrict__ sd, const float* __restrict__ mx,
                          float* __restrict__ ebuf, float* __restrict__ sm) {
  long e = (long)blockIdx.x * blockDim.x + threadIdx.x;
  if (e < (long)EL) {
    int s, d; edge_sd(src, dst, e, s, d);
    float ev = expf(lrelu(ss[s] + sd[d]) - mx[d]);
    ebuf[e] = ev;
    atomicAdd(&sm[d], ev);
  }
}
__global__ void k_aggregate(const int* src, const int* dst, const float* __restrict__ ebuf,
                            const float* __restrict__ sm, const float* __restrict__ Hm,
                            float* __restrict__ OUT, float* alpha_out, int F) {
  long e = blockIdx.x;
  int s, d; edge_sd(src, dst, e, s, d);
  float alpha = ebuf[e] / sm[d];
  if (alpha_out && threadIdx.x == 0) alpha_out[e] = alpha;
  const float* hs = Hm + (size_t)s * F;
  float* od = OUT + (size_t)d * F;
  for (int c = threadIdx.x; c < F; c += blockDim.x) atomicAdd(&od[c], alpha * hs[c]);
}
__global__ void k_relu_bias_bf(const float* __restrict__ OUT, const float* __restrict__ b,
                               unsigned short* __restrict__ hbf, int F, long total) {
  long i = (long)blockIdx.x * blockDim.x + threadIdx.x;
  if (i < total) { float v = OUT[i] + b[i % F]; hbf[i] = f2bf(v > 0.f ? v : 0.f); }
}
// mean-pool relu(OUT2 + b2) into d_out[0..255]; 64 rows per block, one column per thread.
__global__ void k_pool(const float* __restrict__ OUT2, const float* __restrict__ b2,
                       float* __restrict__ dout) {
  int c = threadIdx.x;                // blockDim.x == OO
  long n0 = (long)blockIdx.x * 64;
  float sum = 0.f;
  for (int r = 0; r < 64; ++r) {
    long n = n0 + r;
    if (n < NN) { float v = OUT2[n * OO + c] + b2[c]; sum += v > 0.f ? v : 0.f; }
  }
  atomicAdd(&dout[c], sum * (1.0f / NN));
}

#define CEILDIV(a, b) (((a) + (b) - 1) / (b))

extern "C" void kernel_launch(void* const* d_in, const int* in_sizes, int n_in,
                              void* d_out, int out_size, void* d_ws, size_t ws_size,
                              hipStream_t stream) {
  (void)in_sizes; (void)n_in; (void)out_size; (void)ws_size;
  const float* x       = (const float*)d_in[0];
  const int*   ei      = (const int*)d_in[1];
  const int*   etype   = (const int*)d_in[2];
  const int*   ntype   = (const int*)d_in[3];
  const float* emb_e   = (const float*)d_in[4];
  const float* emb_n   = (const float*)d_in[5];
  const float* fc_w    = (const float*)d_in[6];
  const float* fc_b    = (const float*)d_in[7];
  const float* g1_w    = (const float*)d_in[8];
  const float* g1_as   = (const float*)d_in[9];
  const float* g1_ad   = (const float*)d_in[10];
  const float* g1_b    = (const float*)d_in[11];
  const float* g2_w    = (const float*)d_in[12];
  const float* g2_as   = (const float*)d_in[13];
  const float* g2_ad   = (const float*)d_in[14];
  const float* g2_b    = (const float*)d_in[15];
  const int* src = ei;
  const int* dst = ei + EE;
  float* dout = (float*)d_out;
  float* alpha2 = dout + OO;

  // ---- carve workspace ----
  char* w = (char*)d_ws;
  auto carve = [&](size_t bytes) { char* p = w; w += (bytes + 255) & ~(size_t)255; return p; };
  unsigned short* nodeBf = (unsigned short*)carve((size_t)NN * CC * 2);
  unsigned short* WtopBf = (unsigned short*)carve((size_t)CC * CC * 2);
  unsigned short* W1Bf   = (unsigned short*)carve((size_t)CC * HH * 2);
  unsigned short* W2Bf   = (unsigned short*)carve((size_t)HH * OO * 2);
  float* ebt   = (float*)carve((size_t)RR * CC * 4);
  float* Y1n   = (float*)carve((size_t)NN * CC * 4);
  float* accum = (float*)carve((size_t)NN * CC * 4);
  float* deg   = (float*)carve((size_t)NN * 4);
  unsigned short* xnBf = (unsigned short*)carve((size_t)NN * CC * 2);
  float* H1    = (float*)carve((size_t)NN * HH * 4);
  float* OUT1  = (float*)carve((size_t)NN * HH * 4);
  unsigned short* h1Bf = (unsigned short*)carve((size_t)NN * HH * 2);
  float* H2    = (float*)carve((size_t)NN * OO * 4);
  float* OUT2  = (float*)carve((size_t)NN * OO * 4);
  float* ss1 = (float*)carve((size_t)NN * 4);
  float* sd1 = (float*)carve((size_t)NN * 4);
  float* mx1 = (float*)carve((size_t)NN * 4);
  float* sm1 = (float*)carve((size_t)NN * 4);
  float* ss2 = (float*)carve((size_t)NN * 4);
  float* sd2 = (float*)carve((size_t)NN * 4);
  float* mx2 = (float*)carve((size_t)NN * 4);
  float* sm2 = (float*)carve((size_t)NN * 4);
  float* ebuf = (float*)carve((size_t)EL * 4);

  const float NEG_INF = -__builtin_inff();
  const long NC = (long)NN * CC, NH = (long)NN * HH, NO = (long)NN * OO;

  // ---- phase 0: inits + weight packing ----
  k_fill<<<CEILDIV(NC, 256), 256, 0, stream>>>(accum, NC, 0.f);
  k_fill<<<CEILDIV(NN, 256), 256, 0, stream>>>(deg, NN, 0.f);
  k_pack_b<<<CEILDIV((long)CC * CC, 256), 256, 0, stream>>>(fc_w, WtopBf, CC, CC);
  k_pack_b<<<CEILDIV((long)CC * HH, 256), 256, 0, stream>>>(g1_w, W1Bf, CC, HH);
  k_pack_b<<<CEILDIV((long)HH * OO, 256), 256, 0, stream>>>(g2_w, W2Bf, HH, OO);
  k_ebt<<<CEILDIV(RR * CC, 256), 256, 0, stream>>>(emb_e, fc_w, fc_b, ebt);
  k_node_prep<<<CEILDIV(NC, 256), 256, 0, stream>>>(x, ntype, emb_n, nodeBf);

  const int MT = NN / 16;  // 3125 M-tiles
  const size_t smem768 = (size_t)4 * (CC / 32) * 1024;  // 98304 B
  const size_t smem256 = (size_t)4 * (HH / 32) * 1024;  // 32768 B
  // ---- phase 1: per-node edge-MLP GEMM  Y1n = relu(node_emb) @ W_top ----
  k_wmma_gemm<<<dim3(CEILDIV(MT, 4), CC / 64), 128, smem768, stream>>>(nodeBf, WtopBf, Y1n, NN, CC, CC);
  // ---- phase 2: scatter-mean onto dst ----
  k_scatter_he<<<EE, 256, 0, stream>>>(src, dst, etype, Y1n, ebt, accum, deg);
  k_xn<<<CEILDIV(NC, 256), 256, 0, stream>>>(accum, deg, xnBf);

  // ---- layer 1: H1 = xn @ W1 ----
  k_wmma_gemm<<<dim3(CEILDIV(MT, 4), HH / 64), 128, smem768, stream>>>(xnBf, W1Bf, H1, NN, CC, HH);
  k_scores<<<CEILDIV(NN, 256), 256, 0, stream>>>(H1, g1_as, g1_ad, ss1, sd1, HH);
  k_fill<<<CEILDIV(NN, 256), 256, 0, stream>>>(mx1, NN, NEG_INF);
  k_fill<<<CEILDIV(NN, 256), 256, 0, stream>>>(sm1, NN, 0.f);
  k_fill<<<CEILDIV(NH, 256), 256, 0, stream>>>(OUT1, NH, 0.f);
  k_logit_max<<<CEILDIV((long)EL, 256), 256, 0, stream>>>(src, dst, ss1, sd1, mx1);
  k_exp_sum<<<CEILDIV((long)EL, 256), 256, 0, stream>>>(src, dst, ss1, sd1, mx1, ebuf, sm1);
  k_aggregate<<<EL, 256, 0, stream>>>(src, dst, ebuf, sm1, H1, OUT1, nullptr, HH);
  k_relu_bias_bf<<<CEILDIV(NH, 256), 256, 0, stream>>>(OUT1, g1_b, h1Bf, HH, NH);

  // ---- layer 2: H2 = h1 @ W2 ----
  k_wmma_gemm<<<dim3(CEILDIV(MT, 4), OO / 64), 128, smem256, stream>>>(h1Bf, W2Bf, H2, NN, HH, OO);
  k_scores<<<CEILDIV(NN, 256), 256, 0, stream>>>(H2, g2_as, g2_ad, ss2, sd2, OO);
  k_fill<<<CEILDIV(NN, 256), 256, 0, stream>>>(mx2, NN, NEG_INF);
  k_fill<<<CEILDIV(NN, 256), 256, 0, stream>>>(sm2, NN, 0.f);
  k_fill<<<CEILDIV(NO, 256), 256, 0, stream>>>(OUT2, NO, 0.f);
  k_fill<<<1, 256, 0, stream>>>(dout, OO, 0.f);
  k_logit_max<<<CEILDIV((long)EL, 256), 256, 0, stream>>>(src, dst, ss2, sd2, mx2);
  k_exp_sum<<<CEILDIV((long)EL, 256), 256, 0, stream>>>(src, dst, ss2, sd2, mx2, ebuf, sm2);
  k_aggregate<<<EL, 256, 0, stream>>>(src, dst, ebuf, sm2, H2, OUT2, alpha2, OO);
  k_pool<<<CEILDIV(NN, 64), OO, 0, stream>>>(OUT2, g2_b, dout);
}
